// CoverageModel2_28509992911037
// MI455X (gfx1250) — matrix operand order repaired and checked by reading, not compiled, not run on hardware
//
#include <hip/hip_runtime.h>

typedef __attribute__((ext_vector_type(16))) _Float16 v16h;
typedef __attribute__((ext_vector_type(8)))  _Float16 v8h;
typedef __attribute__((ext_vector_type(8)))  float    v8f;
typedef __attribute__((ext_vector_type(4)))  float    v4f;
typedef __attribute__((ext_vector_type(4)))  unsigned int u32x4;
typedef __attribute__((ext_vector_type(8)))  int      i32x8;
typedef __attribute__((ext_vector_type(4)))  int      i32x4;

#define DEVINL __device__ __forceinline__

// ---------------- LDS layout (bytes), all 16B-aligned ----------------------
#define ACT_A_OFF   0u              // 8 waves x 16 rows x 264 halves
#define ACT_B_OFF   67584u          // 8 waves x 16 rows x 136 halves
#define W2L_OFF     102400u         // 128 rows x 264 halves (256 + 8 pad)
#define W3L_OFF     169984u         //  64 rows x  72 halves ( 64 + 8 pad)
#define W4L_OFF     179200u         //  32 rows x  40 halves ( 32 + 8 pad)
#define W5L_OFF     181760u         // 512 rows x  40 halves ( 32 + 8 pad)
#define SMEM_TOTAL  222720u         // < 320KB / WGP

// ---------------- d_ws layout (bytes) --------------------------------------
#define WS_NZ   0u
#define WS_W1   16u                         // 131072 halves
#define WS_W2   (WS_W1 + 262144u)           //  32768 halves
#define WS_W3   (WS_W2 + 65536u)            //   8192 halves
#define WS_W4   (WS_W3 + 16384u)            //   2048 halves
#define WS_W5   (WS_W4 + 4096u)             //  16384 halves
#define WS_END  (WS_W5 + 32768u)            // 380944 B total scratch needed

// ---------------------------------------------------------------------------
// CDNA5 WMMA wrapper: D = A(16x32 f16) * B(32x16 f16) + C(16x16 f32)
// ---------------------------------------------------------------------------
DEVINL v8f wmma_f16(v16h a, v16h b, v8f c) {
  return __builtin_amdgcn_wmma_f32_16x16x32_f16(
      false, a, false, b, (short)0, c, false, false);
}

DEVINL v16h cat8(v8h lo, v8h hi) {
  v16h a;
#pragma unroll
  for (int i = 0; i < 8; ++i) { a[i] = lo[i]; a[8 + i] = hi[i]; }
  return a;
}

// A fragment (16x32 f16) straight from global fp32 x [B,512] (ISA 7.12.2):
// lanes 0-15: M=lane, K=kb+0..7 / kb+16..23; lanes 16-31: K chunks +8 / +24.
DEVINL v16h load_a_x(const float* __restrict__ x, int mbase, unsigned lane, int kb) {
  const int m  = mbase + (int)(lane & 15u);
  const int k0 = kb + (int)((lane >> 4) << 3);
  const float* p = x + (size_t)m * 512 + k0;
  v4f f0 = *(const v4f*)(p);
  v4f f1 = *(const v4f*)(p + 4);
  v4f f2 = *(const v4f*)(p + 16);
  v4f f3 = *(const v4f*)(p + 20);
  v16h r;
#pragma unroll
  for (int i = 0; i < 4; ++i) {
    r[i]      = (_Float16)f0[i];
    r[4 + i]  = (_Float16)f1[i];
    r[8 + i]  = (_Float16)f2[i];
    r[12 + i] = (_Float16)f3[i];
  }
  return r;
}

// A fragment from f16 LDS slab (stride in halves; stride*2 % 16 == 0).
DEVINL v16h load_a_lds(const _Float16* base, int stride_h, unsigned lane, int kb) {
  const int m  = (int)(lane & 15u);
  const int k0 = kb + (int)((lane >> 4) << 3);
  const _Float16* p = base + m * stride_h + k0;
  return cat8(*(const v8h*)(p), *(const v8h*)(p + 16));
}

// B fragment (32x16 f16, KxN) from pre-converted f16 weights in GLOBAL
// (row-major [N_out, K]); lanes 0-15: K=kb+0..7 / kb+16..23; lanes 16-31: +8 / +24.
DEVINL v16h load_b_g(const _Float16* __restrict__ w, int ldw, unsigned lane,
                     int nb, int kb) {
  const int n  = nb + (int)(lane & 15u);
  const int k0 = kb + (int)((lane >> 4) << 3);
  const _Float16* p = w + (size_t)n * ldw + k0;
  return cat8(*(const v8h*)(p), *(const v8h*)(p + 16));
}

// B fragment from LDS f16 weight image with padded row stride (bank-friendly).
DEVINL v16h load_b_lds(const _Float16* wl, int stride_h, unsigned lane,
                       int nb, int kb) {
  const int n  = nb + (int)(lane & 15u);
  const int k0 = kb + (int)((lane >> 4) << 3);
  const _Float16* p = wl + n * stride_h + k0;
  return cat8(*(const v8h*)(p), *(const v8h*)(p + 16));
}

DEVINL float sigmoidf_fast(float v) { return 1.0f / (1.0f + __expf(-v)); }

// ---------------------------------------------------------------------------
// Tensor Data Mover: 2D tile (whole f16 matrix) global -> LDS with row
// padding.  D# packed per CDNA5 ISA.
//   d0   = row length in elements (tensor_dim0 == tile_dim0 == stride0)
//   rows = number of rows (tensor_dim1 == tile_dim1)
//   pi   = pad_interval code (2^(pi+1) DWORDs between pads)
//   pa   = pad_amount code (pa+1 DWORDs of pad)  -> 3 == 16 bytes
// ---------------------------------------------------------------------------
DEVINL void tdm_load_2d(unsigned lds_off, const void* gsrc,
                        unsigned d0, unsigned rows, unsigned pi, unsigned pa) {
  const unsigned long long ga = (unsigned long long)gsrc;
  u32x4 g0;
  g0[0] = 1u;                                            // count=1, user D#
  g0[1] = lds_off;                                       // lds_addr
  g0[2] = (unsigned)(ga & 0xFFFFFFFFull);                // global_addr[31:0]
  g0[3] = (unsigned)((ga >> 32) & 0x1FFFFFFull)          // global_addr[56:32]
        | (2u << 30);                                    // type = 2 (image)
  i32x8 g1;
  g1[0] = (int)((1u << 16)                               // data_size = 2B
              | (1u << 20)                               // pad_enable
              | (pi << 22) | (pa << 25));                // pad interval/amount
  g1[1] = (int)((d0 & 0xFFFFu) << 16);                   // tensor_dim0[15:0]
  g1[2] = (int)(((d0 >> 16) & 0xFFFFu) | ((rows & 0xFFFFu) << 16));
  g1[3] = (int)(((rows >> 16) & 0xFFFFu) | ((d0 & 0xFFFFu) << 16)); // tile_dim0
  g1[4] = (int)(rows & 0xFFFFu);                         // tile_dim1 (tile_dim2=0)
  g1[5] = (int)d0;                                       // tensor_dim0_stride
  g1[6] = 0; g1[7] = 0;
  i32x4 gz4 = {};
  i32x8 gz8 = {};
  // 6-arg clang-23 form: (g0, g1, g2, g3, g4, cpol)
  __builtin_amdgcn_tensor_load_to_lds(g0, g1, gz4, gz4, gz8, 0);
}

// ---------------------------------------------------------------------------
// Prep: fp32 -> f16 weight images (once; tiny, L2-resident afterwards)
// ---------------------------------------------------------------------------
__global__ void __launch_bounds__(256)
cvt_f32_f16(const float* __restrict__ src, _Float16* __restrict__ dst, int n) {
  int i = (int)blockIdx.x * 256 + (int)threadIdx.x;
  if (i < n) dst[i] = (_Float16)src[i];
}

__global__ void __launch_bounds__(512)
count_nz_row0(const float* __restrict__ x, int* __restrict__ out) {
  __shared__ int cnt;
  if (threadIdx.x == 0) cnt = 0;
  __syncthreads();
  if (x[threadIdx.x] != 0.0f) atomicAdd(&cnt, 1);
  __syncthreads();
  if (threadIdx.x == 0) out[0] = cnt;
}

// ---------------------------------------------------------------------------
// Hidden layer from wave-private LDS activations, LDS-resident weights.
// ---------------------------------------------------------------------------
template <int KDIM, int NDIM>
DEVINL void mlp_layer(const _Float16* inb, int in_str,
                      const _Float16* wl, int w_str,
                      const float* __restrict__ bias,
                      _Float16* outb, int out_str, unsigned lane) {
  constexpr int NT = NDIM / 16;
  constexpr int KT = KDIM / 32;
  v8f acc[NT] = {};
  for (int k = 0; k < KT; ++k) {
    v16h a = load_a_lds(inb, in_str, lane, k * 32);
#pragma unroll
    for (int n = 0; n < NT; ++n) {
      v16h b = load_b_lds(wl, w_str, lane, n * 16, k * 32);
      acc[n] = wmma_f16(a, b, acc[n]);
    }
  }
  const int rbase = (int)((lane >> 4) << 3);
  const int col   = (int)(lane & 15u);
#pragma unroll
  for (int n = 0; n < NT; ++n) {
    const float bv = bias[n * 16 + col];
#pragma unroll
    for (int r = 0; r < 8; ++r) {
      outb[(rbase + r) * out_str + n * 16 + col] =
          (_Float16)sigmoidf_fast(acc[n][r] + bv);
    }
  }
}

// ---------------------------------------------------------------------------
// Fused MLP: 512 blocks x 256 threads (8 wave32); each wave owns 16 batch
// rows.  TDM streams w2..w5 (f16) into LDS while layer 1 computes.
// ---------------------------------------------------------------------------
__global__ void __launch_bounds__(256)
mlp_fused(const float* __restrict__ x,
          const _Float16* __restrict__ w1h, const _Float16* __restrict__ w2h,
          const _Float16* __restrict__ w3h, const _Float16* __restrict__ w4h,
          const _Float16* __restrict__ w5h,
          const float* __restrict__ b1, const float* __restrict__ b2,
          const float* __restrict__ b3, const float* __restrict__ b4,
          const float* __restrict__ b5,
          const int* __restrict__ nz, float* __restrict__ out) {
  const unsigned tid  = threadIdx.x;
  const unsigned lane = tid & 31u;
  const unsigned wave = tid >> 5;
  const int mbase = (int)blockIdx.x * 128 + (int)wave * 16;

  extern __shared__ char smem[];
  _Float16* bufA = (_Float16*)(smem + ACT_A_OFF) + wave * (16 * 264);
  _Float16* bufB = (_Float16*)(smem + ACT_B_OFF) + wave * (16 * 136);
  const _Float16* w2l = (const _Float16*)(smem + W2L_OFF);
  const _Float16* w3l = (const _Float16*)(smem + W3L_OFF);
  const _Float16* w4l = (const _Float16*)(smem + W4L_OFF);
  const _Float16* w5l = (const _Float16*)(smem + W5L_OFF);

  const int in_size = nz[0];

  // Kick off TDM weight staging (overlaps the whole layer-1 GEMM).
  // Row pad of 4 DWORDs makes LDS strides 264/72/40/40 halves (bank-spread).
  if (wave == 0) {
    const unsigned base = (unsigned)(unsigned long long)(uintptr_t)smem;
    tdm_load_2d(base + W2L_OFF, w2h, 256u, 128u, 6u, 3u);  // 128dw rows
    tdm_load_2d(base + W3L_OFF, w3h,  64u,  64u, 4u, 3u);  //  32dw rows
    tdm_load_2d(base + W4L_OFF, w4h,  32u,  32u, 3u, 3u);  //  16dw rows
    tdm_load_2d(base + W5L_OFF, w5h,  32u, 512u, 3u, 3u);  //  16dw rows
  }

  __builtin_prefetch(w1h + (size_t)tid * 1024, 0, 1);      // warm L2 with w1

  // ---- Layer 1: x[512] -> bufA[256] (stride 264).  16 accumulators live in
  // VGPRs; each x fragment is fetched from HBM exactly once (K-outer).
  {
    v8f acc[16] = {};
    for (int k = 0; k < 16; ++k) {
      v16h a = load_a_x(x, mbase, lane, k * 32);
#pragma unroll
      for (int n = 0; n < 16; ++n) {
        v16h b = load_b_g(w1h, 512, lane, n * 16, k * 32);
        acc[n] = wmma_f16(a, b, acc[n]);
      }
    }
    const int rbase = (int)((lane >> 4) << 3);
    const int col   = (int)(lane & 15u);
#pragma unroll
    for (int n = 0; n < 16; ++n) {
      const float bv = b1[n * 16 + col];
#pragma unroll
      for (int r = 0; r < 8; ++r) {
        bufA[(rbase + r) * 264 + n * 16 + col] =
            (_Float16)sigmoidf_fast(acc[n][r] + bv);
      }
    }
  }

  // Publish TDM-staged weights to all waves (single barrier in the kernel).
  if (wave == 0) __builtin_amdgcn_s_wait_tensorcnt((short)0);
  __syncthreads();

  // ---- Layers 2-4: LDS activations x LDS weights, no further barriers.
  mlp_layer<256, 128>(bufA, 264, w2l, 264, b2, bufB, 136, lane);
  mlp_layer<128,  64>(bufB, 136, w3l,  72, b3, bufA,  72, lane);
  mlp_layer< 64,  32>(bufA,  72, w4l,  40, b4, bufB,  40, lane);

  // ---- Layer 5: bufB[32] -> out[512], prefix mask, no activation.
  {
    v16h a = load_a_lds(bufB, 40, lane, 0);   // K=32: one fragment
    const int rbase = (int)((lane >> 4) << 3);
    const int col15 = (int)(lane & 15u);
#pragma unroll
    for (int n = 0; n < 32; ++n) {
      v8f accz = {};
      v16h b = load_b_lds(w5l, 40, lane, n * 16, 0);
      v8f y = wmma_f16(a, b, accz);
      const int c = n * 16 + col15;
      const float bv = b5[c];
      const bool keep = (c < in_size);
#pragma unroll
      for (int r = 0; r < 8; ++r) {
        const int row = mbase + rbase + r;
        out[(size_t)row * 512 + c] = keep ? (y[r] + bv) : 0.0f;
      }
    }
  }
}

// ---------------------------------------------------------------------------
extern "C" void kernel_launch(void* const* d_in, const int* in_sizes, int n_in,
                              void* d_out, int out_size, void* d_ws, size_t ws_size,
                              hipStream_t stream) {
  (void)in_sizes; (void)n_in; (void)out_size; (void)ws_size; // needs >= 380944 B
  const float* x  = (const float*)d_in[0];
  const float* w1 = (const float*)d_in[1];
  const float* b1 = (const float*)d_in[2];
  const float* w2 = (const float*)d_in[3];
  const float* b2 = (const float*)d_in[4];
  const float* w3 = (const float*)d_in[5];
  const float* b3 = (const float*)d_in[6];
  const float* w4 = (const float*)d_in[7];
  const float* b4 = (const float*)d_in[8];
  const float* w5 = (const float*)d_in[9];
  const float* b5 = (const float*)d_in[10];
  float* out = (float*)d_out;
  char*  ws  = (char*)d_ws;

  int*      nz  = (int*)(ws + WS_NZ);
  _Float16* w1h = (_Float16*)(ws + WS_W1);
  _Float16* w2h = (_Float16*)(ws + WS_W2);
  _Float16* w3h = (_Float16*)(ws + WS_W3);
  _Float16* w4h = (_Float16*)(ws + WS_W4);
  _Float16* w5h = (_Float16*)(ws + WS_W5);

  count_nz_row0<<<1, 512, 0, stream>>>(x, nz);
  cvt_f32_f16<<<(131072 + 255) / 256, 256, 0, stream>>>(w1, w1h, 131072);
  cvt_f32_f16<<<( 32768 + 255) / 256, 256, 0, stream>>>(w2, w2h,  32768);
  cvt_f32_f16<<<(  8192 + 255) / 256, 256, 0, stream>>>(w3, w3h,   8192);
  cvt_f32_f16<<<(  2048 + 255) / 256, 256, 0, stream>>>(w4, w4h,   2048);
  cvt_f32_f16<<<( 16384 + 255) / 256, 256, 0, stream>>>(w5, w5h,  16384);

  mlp_fused<<<65536 / 128, 256, SMEM_TOTAL, stream>>>(
      x, w1h, w2h, w3h, w4h, w5h, b1, b2, b3, b4, b5, nz, out);
}
